// PredictionHead_46729244180770
// MI455X (gfx1250) — compile-verified, matrix operand order checked by reading, not run on hardware
//
#include <hip/hip_runtime.h>

typedef __attribute__((ext_vector_type(16))) __bf16 v16bf;
typedef __attribute__((ext_vector_type(8)))  float  v8f;

#define C_CH 256
#define K3   2304        // 9*256
#define NK1  (K3 / 32)   // 72 K-steps for GEMM1
#define OC   255
#define TOTROWS 25575    // 3*8525

__device__ __forceinline__ unsigned short f2bf(float f) {
    unsigned int u = __builtin_bit_cast(unsigned int, f);
    u = u + 0x7fffu + ((u >> 16) & 1u);   // RNE
    return (unsigned short)(u >> 16);
}

union Frag { v16bf v; unsigned int d[8]; };

// gfx1250 async direct-to-LDS copy (ASYNCcnt-tracked), 16 bytes per lane.
__device__ __forceinline__ void async_b128_to_lds(unsigned int lds_byte_addr,
                                                  const void* gsrc) {
    asm volatile("global_load_async_to_lds_b128 %0, %1, off"
                 :: "v"(lds_byte_addr), "v"(gsrc) : "memory");
}
__device__ __forceinline__ void wait_async0() {
    asm volatile("s_wait_asynccnt 0x0" ::: "memory");
}

// ---------------------------------------------------------------------------
// Median-based per-tensor scale: blocks 0..4 -> w3 levels, 5..9 -> w1 levels.
// max-reduce, then two 1024-bin histogram refinement rounds targeting rank n/2.
// ---------------------------------------------------------------------------
__global__ __launch_bounds__(256) void scale_kernel(const float* __restrict__ w3,
                                                    const float* __restrict__ w1,
                                                    float* __restrict__ sOut) {
    const int bid = blockIdx.x;
    const int lvl = bid % 5;
    const bool is3 = bid < 5;
    const int n = is3 ? (C_CH * C_CH * 9) : (OC * C_CH);
    const float* base = is3 ? (w3 + (size_t)lvl * (C_CH * C_CH * 9))
                            : (w1 + (size_t)lvl * (OC * C_CH));
    const int tid = threadIdx.x;

    __shared__ unsigned int hist[1024];
    __shared__ float red[256];
    __shared__ float s_lo, s_step;
    __shared__ unsigned int s_below;

    float mx = 0.f;
    for (int i = tid; i < n; i += 256) mx = fmaxf(mx, fabsf(base[i]));
    red[tid] = mx;
    __syncthreads();
    for (int s = 128; s > 0; s >>= 1) {
        if (tid < s) red[tid] = fmaxf(red[tid], red[tid + s]);
        __syncthreads();
    }
    if (tid == 0) {
        s_lo = 0.f;
        s_step = (red[0] * 1.000001f + 1e-30f) * (1.f / 1024.f);
        s_below = 0u;
    }
    __syncthreads();

    for (int round = 0; round < 2; ++round) {
        for (int j = tid; j < 1024; j += 256) hist[j] = 0u;
        __syncthreads();
        const float lo = s_lo, step = s_step, inv = 1.f / step;
        for (int i = tid; i < n; i += 256) {
            float v = fabsf(base[i]);
            int b = (int)floorf((v - lo) * inv);
            if (b >= 0 && b < 1024) atomicAdd(&hist[b], 1u);
        }
        __syncthreads();
        if (tid == 0) {
            unsigned int rank = (unsigned int)(n / 2) - s_below;
            unsigned int cum = 0; int b = 0;
            for (; b < 1024; ++b) {
                if (cum + hist[b] > rank) break;
                cum += hist[b];
            }
            if (b >= 1024) b = 1023;
            s_below += cum;
            s_lo = lo + b * step;
            s_step = step * (1.f / 1024.f);
            if (round == 1) sOut[bid] = fmaxf(s_lo + 0.5f * step, 1e-5f);
        }
        __syncthreads();
    }
}

// ---------------------------------------------------------------------------
// Ternary quantize + transpose w3 (L,O,I,3,3) -> W3T[lvl][o][k], k = tap*256+ci
// ---------------------------------------------------------------------------
__global__ __launch_bounds__(256) void quant3_kernel(const float* __restrict__ w3,
                                                     const float* __restrict__ s,
                                                     unsigned short* __restrict__ W3T) {
    int i = blockIdx.x * 256 + threadIdx.x;
    if (i >= 5 * C_CH * K3) return;
    int k   = i % K3;
    int o   = (i / K3) % C_CH;
    int lvl = i / (K3 * C_CH);
    int r = k >> 8, ci = k & 255;
    float sc = s[lvl];
    float w = w3[(((size_t)(lvl * C_CH + o) * C_CH + ci) * 9) + r];
    float q = rintf(w / sc);
    q = fminf(1.f, fmaxf(-1.f, q)) * sc;
    W3T[i] = f2bf(q);
}

// w1 (L,255,256) -> W1T[lvl][o][ci], N padded to 256 (row 255 = 0)
__global__ __launch_bounds__(256) void quant1_kernel(const float* __restrict__ w1,
                                                     const float* __restrict__ s,
                                                     unsigned short* __restrict__ W1T) {
    int i = blockIdx.x * 256 + threadIdx.x;
    if (i >= 5 * C_CH * C_CH) return;
    int ci  = i & 255;
    int o   = (i >> 8) & 255;
    int lvl = i >> 16;
    float q = 0.f;
    if (o < OC) {
        float sc = s[5 + lvl];
        float w = w1[((size_t)(lvl * OC + o) * C_CH) + ci];
        q = rintf(w / sc);
        q = fminf(1.f, fmaxf(-1.f, q)) * sc;
    }
    W1T[i] = f2bf(q);
}

// ---------------------------------------------------------------------------
// Fused head: implicit-GEMM 3x3 conv (bf16 WMMA, f32 acc) -> BN -> SiLU ->
// 1x1 conv WMMA -> bias -> NHWC concat scatter.
// Block: 256 thr = 8 waves; tile M=64 x N=256; double-buffered K pipeline,
// B tiles staged via global_load_async_to_lds_b128 (ASYNCcnt).
// LDS: aBuf[2] 64x32 bf16 (17-dword rows), bBuf[2] 256x32 bf16 (20-dword
// rows, 16B-aligned async chunks), h 64x256 bf16 aliased over staging.
// ---------------------------------------------------------------------------
__global__ __launch_bounds__(256) void head_kernel(
    const float* __restrict__ feat,
    const unsigned short* __restrict__ W3T,   // [256][2304] bf16, this level
    const unsigned short* __restrict__ W1T,   // [256][256]  bf16, this level
    const float* __restrict__ gamma, const float* __restrict__ beta,
    const float* __restrict__ mean,  const float* __restrict__ var,
    const float* __restrict__ bias,  float* __restrict__ out,
    int H, int W, int HW, int M, int rowoff) {

    __shared__ __align__(16) unsigned int smem[12416];   // 49664 B
    unsigned int* const aB0 = smem;                       // 64*17
    unsigned int* const aB1 = smem + 1088;                // 64*17
    unsigned int* const bB0 = smem + 2176;                // 256*20
    unsigned int* const bB1 = smem + 7296;                // 256*20
    unsigned int* const hU  = smem;                       // 64*129 (aliased)

    const int tid  = threadIdx.x;
    const int lane = tid & 31;
    const int l16  = lane & 15;
    const int kh   = lane >> 4;             // K-half selector per ISA layout
    const int wid  = tid >> 5;
    const int wm   = wid & 3;               // wave M sub-tile (0..3)
    const int wn   = wid >> 2;              // wave N half (0..1)

    // staging identity: fixed m per thread, t varies
    const int mloc = tid & 63;
    const int tgrp = tid >> 6;              // 0..3
    const int m_g  = blockIdx.x * 64 + mloc;
    const bool mvalid = m_g < M;
    int pb = 0, py = 0, px = 0;
    if (mvalid) {
        pb = m_g / HW;
        int rem = m_g - pb * HW;
        py = rem / W;
        px = rem - py * W;
    }
    const size_t featB = (size_t)pb * C_CH * HW;

    v8f acc[8];
#pragma unroll
    for (int t = 0; t < 8; ++t)
#pragma unroll
        for (int v = 0; v < 8; ++v) acc[t][v] = 0.f;

    // ---- staging helpers (inlined via lambdas) ----
    auto stageA = [&](int kb, unsigned int* aN) {
        unsigned short* aH = (unsigned short*)aN;
#pragma unroll
        for (int it = 0; it < 8; ++it) {
            int t = tgrp * 8 + it;
            int k = kb * 32 + t;
            int r = k >> 8, ci = k & 255;
            int r3 = r / 3;
            int dy = r3 - 1, dx = (r - r3 * 3) - 1;
            float v = 0.f;
            int yy = py + dy, xx = px + dx;
            if (mvalid && yy >= 0 && yy < H && xx >= 0 && xx < W)
                v = feat[featB + (size_t)ci * HW + (size_t)yy * W + xx];
            aH[mloc * 34 + t] = f2bf(v);
        }
    };
    auto stageB = [&](int kb, unsigned int* bN) {
        // thread tid copies weight row n = tid: 4 async 16B chunks
        const unsigned short* src = W3T + (size_t)tid * K3 + kb * 32;
        unsigned int dstb = (unsigned int)(size_t)(bN + tid * 20);
#pragma unroll
        for (int c = 0; c < 4; ++c)
            async_b128_to_lds(dstb + c * 16, src + c * 8);
    };

    // ---- prologue: fill buffer 0 ----
    stageA(0, aB0);
    stageB(0, bB0);
    wait_async0();
    __syncthreads();

    // ---- main K pipeline: one barrier per step, async B prefetch ----
    for (int kb = 0; kb < NK1; ++kb) {
        const unsigned int* aC = (kb & 1) ? aB1 : aB0;
        const unsigned int* bC = (kb & 1) ? bB1 : bB0;
        if (kb + 1 < NK1) {
            unsigned int* aN = (kb & 1) ? aB0 : aB1;
            unsigned int* bN = (kb & 1) ? bB0 : bB1;
            stageA(kb + 1, aN);
            stageB(kb + 1, bN);
        }

        Frag a;
        const unsigned int* ar = aC + (wm * 16 + l16) * 17 + kh * 4;
#pragma unroll
        for (int r = 0; r < 4; ++r) { a.d[r] = ar[r]; a.d[4 + r] = ar[8 + r]; }
#pragma unroll
        for (int tt = 0; tt < 8; ++tt) {
            Frag b;
            const unsigned int* br = bC + (wn * 128 + tt * 16 + l16) * 20 + kh * 4;
#pragma unroll
            for (int r = 0; r < 4; ++r) { b.d[r] = br[r]; b.d[4 + r] = br[8 + r]; }
            acc[tt] = __builtin_amdgcn_wmma_f32_16x16x32_bf16(
                false, a.v, false, b.v, (short)0, acc[tt], false, false);
        }

        wait_async0();          // next-tile async B writes complete
        __syncthreads();        // staging visible / safe to overwrite cur
    }

    // ---- epilogue 1: BN + SiLU, stash bf16 h in LDS (aliases staging) ----
    unsigned short* hH = (unsigned short*)hU;
#pragma unroll
    for (int tt = 0; tt < 8; ++tt) {
        int n = wn * 128 + tt * 16 + l16;
        float as = gamma[n] * rsqrtf(var[n] + 1e-5f);
        float bs = beta[n] - mean[n] * as;
#pragma unroll
        for (int v = 0; v < 8; ++v) {
            float h = acc[tt][v] * as + bs;
            h = h / (1.f + __expf(-h));
            int m = wm * 16 + v + 8 * kh;
            hH[m * 258 + n] = f2bf(h);
        }
    }
    __syncthreads();

    // ---- GEMM2: 1x1 conv, K = 256 (8 WMMA K-steps), B from L2 ----
    v8f acc2[8];
#pragma unroll
    for (int t = 0; t < 8; ++t)
#pragma unroll
        for (int v = 0; v < 8; ++v) acc2[t][v] = 0.f;

    const unsigned int* w1u = (const unsigned int*)W1T;
    for (int kb2 = 0; kb2 < 8; ++kb2) {
        Frag a2;
        const unsigned int* ar2 = hU + (wm * 16 + l16) * 129 + kb2 * 16 + kh * 4;
#pragma unroll
        for (int r = 0; r < 4; ++r) { a2.d[r] = ar2[r]; a2.d[4 + r] = ar2[8 + r]; }
#pragma unroll
        for (int tt = 0; tt < 8; ++tt) {
            int n = wn * 128 + tt * 16 + l16;
            const unsigned int* br2 = w1u + n * 128 + kb2 * 16 + kh * 4;
            Frag b2;
#pragma unroll
            for (int r = 0; r < 4; ++r) { b2.d[r] = br2[r]; b2.d[4 + r] = br2[8 + r]; }
            acc2[tt] = __builtin_amdgcn_wmma_f32_16x16x32_bf16(
                false, a2.v, false, b2.v, (short)0, acc2[tt], false, false);
        }
    }

    // ---- epilogue 2: + bias, NCHW -> (B, HW*3, 85) concat scatter ----
#pragma unroll
    for (int tt = 0; tt < 8; ++tt) {
        int n = wn * 128 + tt * 16 + l16;
        if (n < OC) {
            float bi = bias[n];
            int p = n / 85;
            int ai = n - p * 85;
#pragma unroll
            for (int v = 0; v < 8; ++v) {
                int mg = blockIdx.x * 64 + wm * 16 + v + 8 * kh;
                if (mg < M) {
                    int bb = mg / HW;
                    int rem = mg - bb * HW;
                    int row = rowoff + rem * 3 + p;
                    out[((size_t)bb * TOTROWS + row) * 85 + ai] = acc2[tt][v] + bi;
                }
            }
        }
    }
}

// ---------------------------------------------------------------------------
extern "C" void kernel_launch(void* const* d_in, const int* in_sizes, int n_in,
                              void* d_out, int out_size, void* d_ws, size_t ws_size,
                              hipStream_t stream) {
    const float* feat[5];
    for (int i = 0; i < 5; ++i) feat[i] = (const float*)d_in[i];
    const float* w3    = (const float*)d_in[5];
    const float* gamma = (const float*)d_in[6];
    const float* beta  = (const float*)d_in[7];
    const float* mean  = (const float*)d_in[8];
    const float* var   = (const float*)d_in[9];
    const float* w1    = (const float*)d_in[10];
    const float* b1    = (const float*)d_in[11];
    float* out = (float*)d_out;

    float* sBuf = (float*)d_ws;                                   // 10 floats
    unsigned short* W3T = (unsigned short*)((char*)d_ws + 64);    // 5*256*2304
    unsigned short* W1T = W3T + (size_t)5 * C_CH * K3;            // 5*256*256

    scale_kernel<<<10, 256, 0, stream>>>(w3, w1, sBuf);

    int n3 = 5 * C_CH * K3;
    quant3_kernel<<<(n3 + 255) / 256, 256, 0, stream>>>(w3, sBuf, W3T);
    int n1 = 5 * C_CH * C_CH;
    quant1_kernel<<<(n1 + 255) / 256, 256, 0, stream>>>(w1, sBuf, W1T);

    static const int SZ[5] = {80, 40, 20, 10, 5};
    int rowoff = 0;
    for (int l = 0; l < 5; ++l) {
        int H = SZ[l], W = SZ[l], HW = H * W;
        int M = 16 * HW;
        int blocks = (M + 63) / 64;
        head_kernel<<<blocks, 256, 0, stream>>>(
            feat[l],
            W3T + (size_t)l * C_CH * K3,
            W1T + (size_t)l * C_CH * C_CH,
            gamma + l * C_CH, beta + l * C_CH, mean + l * C_CH, var + l * C_CH,
            b1 + l * OC, out, H, W, HW, M, rowoff);
        rowoff += 3 * HW;
    }
}
